// SourceBasedTGNMemory_13769665151521
// MI455X (gfx1250) — compile-verified, matrix operand order checked by reading, not run on hardware
//
#include <hip/hip_runtime.h>
#include <hip/hip_bf16.h>
#include <math.h>

#define N_NODES 200000
#define MEM_DIM 256
#define MSG_DIM 256
#define BATCH   2048
#define NSLOT   (2 * BATCH)
#define KPAD1   544           // 515 padded up to multiple of 32

typedef __attribute__((ext_vector_type(16))) __bf16 v16bf;
typedef __attribute__((ext_vector_type(8)))  float  v8f;

// ---------------------------------------------------------------------------
// WMMA bf16 fragment loader. 16-bit A-matrix 16x32 layout (ISA 7.12.2):
// lane L (r = L&15, half = L>>4) holds row M=r, K = half*8 + {0..7} in elems
// 0..7 and K = 16 + half*8 + {0..7} in elems 8..15. B (32x16) mirrors with
// N across lanes, so the same loader works on a row-major [N][K] weight.
// ---------------------------------------------------------------------------
__device__ inline v16bf frag_ld(const __bf16* base, int ld, int row0, int k0) {
  int lane = threadIdx.x & 31;
  int r    = lane & 15;
  int half = lane >> 4;
  const __bf16* q = base + (size_t)(row0 + r) * ld + k0 + half * 8;
  v16bf f;
#pragma unroll
  for (int e = 0; e < 8; ++e) { f[e] = q[e]; f[e + 8] = q[e + 16]; }
  return f;
}

// Per-lane LDS byte address for async global->LDS copies. Generic (flat)
// pointers to __shared__ carry the LDS offset in their low 32 bits
// (ISA 10.2 aperture mapping).
__device__ inline unsigned lds_addr32(const void* p) {
  return (unsigned)(unsigned long long)p;
}

// ---------------------------------------------------------------------------
// f32 -> bf16 weight conversion with zero K-padding.
// ---------------------------------------------------------------------------
__global__ void cvt_pad_kernel(const float* __restrict__ src, __bf16* __restrict__ dst,
                               int N, int Ks, int Kd) {
  int idx = blockIdx.x * blockDim.x + threadIdx.x;
  if (idx >= N * Kd) return;
  int n = idx / Kd, k = idx - n * Kd;
  dst[idx] = (__bf16)(k < Ks ? src[(size_t)n * Ks + k] : 0.0f);
}

// Build x = [src_mem | dst_mem | edge | 0-pad] as bf16 (2048 x 544).
__global__ void build_x_kernel(const float* __restrict__ memory,
                               const int* __restrict__ src, const int* __restrict__ dst,
                               const float* __restrict__ edge, __bf16* __restrict__ xbuf) {
  int idx = blockIdx.x * blockDim.x + threadIdx.x;
  if (idx >= BATCH * KPAD1) return;
  int b = idx / KPAD1, k = idx - b * KPAD1;
  float v;
  if      (k < 256) v = memory[(size_t)src[b] * MEM_DIM + k];
  else if (k < 512) v = memory[(size_t)dst[b] * MEM_DIM + (k - 256)];
  else if (k < 515) v = edge[b * 3 + (k - 512)];
  else              v = 0.0f;
  xbuf[idx] = (__bf16)v;
}

__global__ void init_kernel(int* head_count) { *head_count = 0; }

// ---------------------------------------------------------------------------
// WMMA GEMM with double-buffered async global->LDS staging:
//   C = act(A(MxK) * B(NxK)^T + bias);  mode 0: f32 out, mode 1: relu+bf16 out.
// Block = 256 threads = 8 waves; block tile 32(M) x 64(N); K chunk = 32.
// Each K-chunk: every thread issues exactly 2 async copies (A: b64, B: b128),
// so `s_wait_asynccnt 2` retires the previous chunk while the next streams in.
// ---------------------------------------------------------------------------
__global__ __launch_bounds__(256)
void gemm_bf16_kernel(const __bf16* __restrict__ A, const __bf16* __restrict__ B,
                      const float* __restrict__ bias, float* __restrict__ Cf,
                      __bf16* __restrict__ Cb, int M, int N, int K, int mode) {
  __shared__ alignas(16) __bf16 As[2][32 * 32];   // 2 x 2 KB
  __shared__ alignas(16) __bf16 Bs[2][64 * 32];   // 2 x 4 KB
  const int tid  = threadIdx.x;
  const int w    = tid >> 5;
  const int lane = tid & 31;
  const int m0   = blockIdx.x * 32;
  const int n0   = blockIdx.y * 64;

  // A stage: 8 bytes (4 bf16)/thread; B stage: 16 bytes (8 bf16)/thread.
  const int ar = tid >> 3, ac = (tid & 7) * 4;
  const int br = tid >> 2, bc = (tid & 3) * 8;

  auto issue = [&](int k0, int buf) {
    unsigned la = lds_addr32(&As[buf][ar * 32 + ac]);
    unsigned lb = lds_addr32(&Bs[buf][br * 32 + bc]);
    unsigned long long ga =
        (unsigned long long)(const void*)(A + (size_t)(m0 + ar) * K + k0 + ac);
    unsigned long long gb =
        (unsigned long long)(const void*)(B + (size_t)(n0 + br) * K + k0 + bc);
    asm volatile("global_load_async_to_lds_b64 %0, %1, off"
                 :: "v"(la), "v"(ga) : "memory");
    asm volatile("global_load_async_to_lds_b128 %0, %1, off"
                 :: "v"(lb), "v"(gb) : "memory");
  };

  const int nch = K / 32;
  issue(0, 0);
  v8f acc = {};
  for (int i = 0; i < nch; ++i) {
    if (i + 1 < nch) {
      issue((i + 1) * 32, (i + 1) & 1);
      asm volatile("s_wait_asynccnt 0x2" ::: "memory");  // chunk i landed
    } else {
      asm volatile("s_wait_asynccnt 0x0" ::: "memory");
    }
    __syncthreads();
    v16bf a = frag_ld(&As[i & 1][0], 32, (w & 1) * 16, 0);
    v16bf b = frag_ld(&Bs[i & 1][0], 32, (w >> 1) * 16, 0);
    acc = __builtin_amdgcn_wmma_f32_16x16x32_bf16(false, a, false, b,
                                                  (short)0, acc, false, false);
    __syncthreads();   // everyone done reading buf before it is rewritten
  }

  int col  = n0 + (w >> 1) * 16 + (lane & 15);
  float bv = bias[col];
#pragma unroll
  for (int r = 0; r < 8; ++r) {
    int row = m0 + (w & 1) * 16 + r + 8 * (lane >> 4);
    float v = acc[r] + bv;
    if (mode == 1) { v = v > 0.0f ? v : 0.0f; Cb[(size_t)row * N + col] = (__bf16)v; }
    else           { Cf[(size_t)row * N + col] = v; }
  }
}

// ---------------------------------------------------------------------------
// Per-slot node ids + chain heads (rank-0 slots). Slot i = event i>>1,
// role i&1 (0=src, 1=dst) — the exact sequential order of the reference scan.
// ---------------------------------------------------------------------------
__global__ void rank_kernel(const int* __restrict__ src, const int* __restrict__ dst,
                            int* __restrict__ slot_node, int* __restrict__ head_list,
                            int* __restrict__ head_count) {
  int i = blockIdx.x * blockDim.x + threadIdx.x;
  if (i >= NSLOT) return;
  int node = (i & 1) ? dst[i >> 1] : src[i >> 1];
  slot_node[i] = node;
  int rank = 0;
  for (int j = 0; j < i; ++j) {
    int nj = (j & 1) ? dst[j >> 1] : src[j >> 1];
    rank += (nj == node);
  }
  if (rank == 0) {
    int pos = atomicAdd(head_count, 1);
    head_list[pos] = i;
  }
}

// ---------------------------------------------------------------------------
// Chain GRU: 16 independent node-chains per block. Each step computes
//   gih = msg @ Wih^T,  ghh = h @ Whh^T   (16x768, bf16 WMMA, f32 acc)
// then f32 elementwise gates; chains advance to the next slot of their node.
// ---------------------------------------------------------------------------
__global__ __launch_bounds__(256)
void gru_chain_kernel(const int* __restrict__ head_list, const int* __restrict__ head_count,
                      const int* __restrict__ slot_node, const float* __restrict__ messages,
                      const float* __restrict__ memory, const float* __restrict__ ts,
                      const __bf16* __restrict__ wihb, const __bf16* __restrict__ whhb,
                      const float* __restrict__ bih, const float* __restrict__ bhh,
                      float* __restrict__ out_mem, float* __restrict__ out_lu) {
  extern __shared__ char smem[];
  float*  gih  = (float*)smem;                       // 16 x 768
  float*  ghh  = gih + 16 * 768;                     // 16 x 768
  __bf16* msgA = (__bf16*)(ghh + 16 * 768);          // 16 x 256
  __bf16* hA   = msgA + 16 * 256;                    // 16 x 256
  int*    cur   = (int*)(hA + 16 * 256);             // [16]
  int*    act   = cur + 16;                          // [16]
  int*    vnode = act + 16;                          // [16]
  float*  tst   = (float*)(vnode + 16);              // [16]
  int*    anyact = (int*)(tst + 16);                 // [1]

  const int tid  = threadIdx.x;
  const int c    = tid >> 4;       // chain 0..15
  const int part = tid & 15;       // 16 dims each (part*16 .. part*16+15)
  const int nheads = *head_count;
  const int hidx   = blockIdx.x * 16 + c;
  const bool valid = hidx < nheads;

  if (part == 0) {
    int slot = valid ? head_list[hidx] : -1;
    cur[c]   = slot;
    act[c]   = valid ? 1 : 0;
    vnode[c] = valid ? slot_node[slot] : -1;
    tst[c]   = 0.0f;
  }
  if (tid == 0) anyact[0] = 1;
  __syncthreads();

  if (blockIdx.x * 16 >= nheads) return;   // block-uniform

  const int v = vnode[c];
  float hreg[16];
#pragma unroll
  for (int e = 0; e < 16; ++e)
    hreg[e] = valid ? memory[(size_t)v * MEM_DIM + part * 16 + e] : 0.0f;

  for (int iter = 0; iter < NSLOT; ++iter) {
    if (!anyact[0]) break;                 // uniform across block
    const bool on = act[c] != 0;
    const int  ms = on ? (cur[c] >> 1) : 0;
#pragma unroll
    for (int e = 0; e < 16; ++e) {
      int j = part * 16 + e;
      msgA[c * 256 + j] = (__bf16)(on ? messages[(size_t)ms * MSG_DIM + j] : 0.0f);
      hA[c * 256 + j]   = (__bf16)(on ? hreg[e] : 0.0f);
    }
    __syncthreads();

    // ---- 16x768 = (16x256)x(256x768)^T, twice, via WMMA -----------------
    const int w = tid >> 5, lane = tid & 31;
    for (int t = 0; t < 6; ++t) {
      const int nb = (w * 6 + t) * 16;
      // weights are streamed from L2 every step: hint the prefetcher
      __builtin_prefetch(wihb + (size_t)nb * 256, 0, 1);
      __builtin_prefetch(whhb + (size_t)nb * 256, 0, 1);
      v8f acc = {};
#pragma unroll
      for (int k0 = 0; k0 < 256; k0 += 32) {
        v16bf a = frag_ld(msgA, 256, 0, k0);
        v16bf b = frag_ld(wihb, 256, nb, k0);
        acc = __builtin_amdgcn_wmma_f32_16x16x32_bf16(false, a, false, b,
                                                      (short)0, acc, false, false);
      }
#pragma unroll
      for (int r = 0; r < 8; ++r)
        gih[(r + 8 * (lane >> 4)) * 768 + nb + (lane & 15)] = acc[r];

      v8f acc2 = {};
#pragma unroll
      for (int k0 = 0; k0 < 256; k0 += 32) {
        v16bf a = frag_ld(hA, 256, 0, k0);
        v16bf b = frag_ld(whhb, 256, nb, k0);
        acc2 = __builtin_amdgcn_wmma_f32_16x16x32_bf16(false, a, false, b,
                                                       (short)0, acc2, false, false);
      }
#pragma unroll
      for (int r = 0; r < 8; ++r)
        ghh[(r + 8 * (lane >> 4)) * 768 + nb + (lane & 15)] = acc2[r];
    }
    __syncthreads();

    // ---- elementwise GRU gates (f32) ------------------------------------
    if (on) {
#pragma unroll
      for (int e = 0; e < 16; ++e) {
        int j = part * 16 + e;
        float ir = gih[c * 768 +       j] + bih[      j];
        float hr = ghh[c * 768 +       j] + bhh[      j];
        float iz = gih[c * 768 + 256 + j] + bih[256 + j];
        float hz = ghh[c * 768 + 256 + j] + bhh[256 + j];
        float in_ = gih[c * 768 + 512 + j] + bih[512 + j];
        float hn  = ghh[c * 768 + 512 + j] + bhh[512 + j];
        float r = 1.0f / (1.0f + __expf(-(ir + hr)));
        float z = 1.0f / (1.0f + __expf(-(iz + hz)));
        float n = tanhf(in_ + r * hn);
        hreg[e] = (1.0f - z) * n + z * hreg[e];
      }
    }
    __syncthreads();

    // ---- advance each chain to its node's next slot ---------------------
    if (part == 0 && act[c]) {
      int s = cur[c];
      tst[c] = ts[s >> 1];
      int nxt = -1;
      for (int q = s + 1; q < NSLOT; ++q)
        if (slot_node[q] == vnode[c]) { nxt = q; break; }
      if (nxt < 0) act[c] = 0; else cur[c] = nxt;
    }
    __syncthreads();
    if (tid == 0) {
      int a = 0;
      for (int q = 0; q < 16; ++q) a |= act[q];
      anyact[0] = a;
    }
    __syncthreads();
  }

  if (valid) {
#pragma unroll
    for (int e = 0; e < 16; ++e)
      out_mem[(size_t)v * MEM_DIM + part * 16 + e] = hreg[e];
    if (part == 0) out_lu[v] = tst[c];
  }
}

// ---------------------------------------------------------------------------
extern "C" void kernel_launch(void* const* d_in, const int* in_sizes, int n_in,
                              void* d_out, int out_size, void* d_ws, size_t ws_size,
                              hipStream_t stream) {
  (void)in_sizes; (void)n_in; (void)out_size; (void)ws_size;
  const int*   src    = (const int*)  d_in[0];
  const int*   dst    = (const int*)  d_in[1];
  const float* edge   = (const float*)d_in[2];
  const float* ts     = (const float*)d_in[3];
  const float* memory = (const float*)d_in[4];
  const float* lastu  = (const float*)d_in[5];
  const float* W1     = (const float*)d_in[6];
  const float* b1     = (const float*)d_in[7];
  const float* W2     = (const float*)d_in[8];
  const float* b2     = (const float*)d_in[9];
  const float* wih    = (const float*)d_in[10];
  const float* whh    = (const float*)d_in[11];
  const float* bih    = (const float*)d_in[12];
  const float* bhh    = (const float*)d_in[13];

  float* out_mem = (float*)d_out;
  float* out_lu  = out_mem + (size_t)N_NODES * MEM_DIM;

  // workspace carve-up (256B aligned)
  char* ws = (char*)d_ws;
  size_t off = 0;
  auto carve = [&](size_t bytes) {
    void* p = ws + off;
    off = (off + bytes + 255) & ~(size_t)255;
    return p;
  };
  __bf16* W1b  = (__bf16*)carve((size_t)MSG_DIM * KPAD1 * 2);
  __bf16* W2b  = (__bf16*)carve((size_t)MSG_DIM * MSG_DIM * 2);
  __bf16* wihb = (__bf16*)carve((size_t)3 * MEM_DIM * MSG_DIM * 2);
  __bf16* whhb = (__bf16*)carve((size_t)3 * MEM_DIM * MEM_DIM * 2);
  __bf16* xbuf = (__bf16*)carve((size_t)BATCH * KPAD1 * 2);
  __bf16* hbuf = (__bf16*)carve((size_t)BATCH * MSG_DIM * 2);
  float*  msg  = (float*) carve((size_t)BATCH * MSG_DIM * 4);
  int* slot_node  = (int*)carve(NSLOT * 4);
  int* head_list  = (int*)carve(NSLOT * 4);
  int* head_count = (int*)carve(4);

  // 1) pass-through copy of memory table + last_update (bandwidth floor)
  hipMemcpyAsync(out_mem, memory, (size_t)N_NODES * MEM_DIM * sizeof(float),
                 hipMemcpyDeviceToDevice, stream);
  hipMemcpyAsync(out_lu, lastu, (size_t)N_NODES * sizeof(float),
                 hipMemcpyDeviceToDevice, stream);

  init_kernel<<<1, 1, 0, stream>>>(head_count);

  // 2) bf16 weight staging
  cvt_pad_kernel<<<(MSG_DIM * KPAD1 + 255) / 256, 256, 0, stream>>>(W1, W1b, MSG_DIM, 515, KPAD1);
  cvt_pad_kernel<<<(MSG_DIM * MSG_DIM + 255) / 256, 256, 0, stream>>>(W2, W2b, MSG_DIM, MSG_DIM, MSG_DIM);
  cvt_pad_kernel<<<(3 * MEM_DIM * MSG_DIM + 255) / 256, 256, 0, stream>>>(wih, wihb, 3 * MEM_DIM, MSG_DIM, MSG_DIM);
  cvt_pad_kernel<<<(3 * MEM_DIM * MEM_DIM + 255) / 256, 256, 0, stream>>>(whh, whhb, 3 * MEM_DIM, MEM_DIM, MEM_DIM);

  // 3) message MLP: x -> relu(x W1^T + b1) -> (.. W2^T + b2)
  build_x_kernel<<<(BATCH * KPAD1 + 255) / 256, 256, 0, stream>>>(memory, src, dst, edge, xbuf);
  gemm_bf16_kernel<<<dim3(BATCH / 32, MSG_DIM / 64), 256, 0, stream>>>(
      xbuf, W1b, b1, nullptr, hbuf, BATCH, MSG_DIM, KPAD1, 1);
  gemm_bf16_kernel<<<dim3(BATCH / 32, MSG_DIM / 64), 256, 0, stream>>>(
      hbuf, W2b, b2, msg, nullptr, BATCH, MSG_DIM, MSG_DIM, 0);

  // 4) chain decomposition of the sequential scan
  rank_kernel<<<(NSLOT + 255) / 256, 256, 0, stream>>>(src, dst, slot_node, head_list, head_count);

  // 5) per-node GRU chains, 16 per block, WMMA gates
  size_t shmem = (size_t)16 * 768 * 4 * 2   // gih + ghh (f32)
               + (size_t)16 * 256 * 2 * 2   // msgA + hA (bf16)
               + 512;                       // chain metadata
  gru_chain_kernel<<<NSLOT / 16, 256, shmem, stream>>>(
      head_list, head_count, slot_node, msg, memory, ts,
      wihb, whhb, bih, bhh, out_mem, out_lu);
}
